// PointNet_SD_Cascade_38208029065494
// MI455X (gfx1250) — matrix operand order repaired and checked by reading, not compiled, not run on hardware
//
#include <hip/hip_runtime.h>
#include <hip/hip_bf16.h>

typedef __attribute__((ext_vector_type(16))) _Float16 v16h;
typedef __attribute__((ext_vector_type(8)))  _Float16 v8h;
typedef __attribute__((ext_vector_type(8)))  float    v8f;

// ---------------------------------------------------------------------------
// WMMA fragment helpers (CDNA5 wave32 layouts, 16x16x32 f16 -> f32)
// A 16x32 f16 tile from LDS (row-major, lda in halves, lda % 32 == 0):
//   lanes 0-15:  row=lane,    halves t=0..15 -> K = kb+t
//   lanes 16-31: row=lane-16, halves t=0..15 -> K = kb+16+t
// ---------------------------------------------------------------------------
__device__ inline v16h frag_a_lds(const _Float16* A, int lda, int row0, int kbase) {
  const int lane = threadIdx.x & 31;
  const int r = lane & 15;
  const int koff = (lane < 16) ? 0 : 16;
  const _Float16* p = A + (size_t)(row0 + r) * lda + kbase + koff;
  const v8h a0 = *reinterpret_cast<const v8h*>(p);        // K = +0..7
  const v8h a1 = *reinterpret_cast<const v8h*>(p + 8);    // K = +8..15
  v16h a;
#pragma unroll
  for (int t = 0; t < 8; ++t) { a[t] = a0[t]; a[8 + t] = a1[t]; }
  return a;
}

// B 32x16 f16 tile from PRE-PACKED weights Wt (Nn x Kpad, col-major-of-W, f16,
// zero-padded K). Per lane: 16 contiguous halves -> one 32B vector load.
//   lanes 0-15:  col = col0+lane,    halves t=0..15 -> K = kbase+t
//   lanes 16-31: col = col0+lane-16, halves t=0..15 -> K = kbase+16+t
// ---------------------------------------------------------------------------
__device__ inline v16h frag_b_pk(const _Float16* __restrict__ Wt, int Kpad,
                                 int kbase, int col0) {
  const int lane = threadIdx.x & 31;
  const int n = col0 + (lane & 15);
  const int kb = kbase + ((lane < 16) ? 0 : 16);
  return *reinterpret_cast<const v16h*>(Wt + (size_t)n * Kpad + kb);
}

// ---------------------------------------------------------------------------
// pack W (K x Nn, f32, row-major) -> Wt (Nn x Kpad, f16), zero pad K..Kpad
// ---------------------------------------------------------------------------
__global__ void k_pack_w(const float* __restrict__ W, _Float16* __restrict__ Wt,
                         int K, int Nn, int Kpad, int total) {
  const int i = blockIdx.x * blockDim.x + threadIdx.x;
  if (i >= total) return;
  const int n = i / Kpad, k = i - n * Kpad;
  Wt[i] = (_Float16)((k < K) ? W[(size_t)k * Nn + n] : 0.0f);
}

// ---------------------------------------------------------------------------
// Fused SA stage: gather(+center) -> MLP layer1(+ReLU) -> layer2 -> max-pool.
// mode 0: kNN grouping, one block per (centroid s, batch b), k rows.
// mode 1: group-all, one block per (16-point group s, batch b), partial pool.
// ---------------------------------------------------------------------------
__global__ __launch_bounds__(128) void k_sa_fused(
    int mode,
    const float* __restrict__ sxyz,   // (B,N,3)
    const float* __restrict__ spts,   // (B,N,Cpts)
    const float* __restrict__ cxyz,   // (B,S,3)   mode 0 only
    const int*   __restrict__ kidx,   // (B,S,k)   mode 0 only
    const _Float16* __restrict__ W1t, const float* __restrict__ b1,
    const _Float16* __restrict__ W2t, const float* __restrict__ b2,
    float* __restrict__ outp,         // (B,S,C2)
    int S, int N, int k, int k_pad, int Cpts, int Cin_pad, int C1, int C2)
{
  extern __shared__ _Float16 sa_sm[];
  _Float16* Ag = sa_sm;                           // k_pad * Cin_pad
  _Float16* Hs = sa_sm + (size_t)k_pad * Cin_pad; // k_pad * C1
  const int b = blockIdx.y, s = blockIdx.x;
  const int tid = threadIdx.x, lane = tid & 31, wv = tid >> 5;
  const int Cin = 3 + Cpts;

  // -- build grouped A tile in LDS (f16), zero-padded rows/cols --------------
  for (int e = tid; e < k_pad * Cin_pad; e += 128) {
    const int j = e / Cin_pad, c = e - j * Cin_pad;
    float v = 0.0f;
    if (j < k && c < Cin) {
      const int p = (mode == 0) ? kidx[((size_t)b * S + s) * k + j] : (s * 16 + j);
      if (c < 3) {
        v = sxyz[((size_t)b * N + p) * 3 + c];
        if (mode == 0) v -= cxyz[((size_t)b * S + s) * 3 + c];
      } else {
        v = spts[((size_t)b * N + p) * Cpts + (c - 3)];
      }
    }
    Ag[e] = (_Float16)v;
  }
  __syncthreads();

  // -- layer 1: Hs = relu(A @ W1 + b1) ---------------------------------------
  const int nct1 = C1 >> 4, nrt = k_pad >> 4;
  for (int ct = wv; ct < nct1; ct += 4) {
    for (int rt = 0; rt < nrt; ++rt) {
      v8f acc = {};
      for (int kb = 0; kb < Cin_pad; kb += 32) {
        v16h a  = frag_a_lds(Ag, Cin_pad, rt * 16, kb);
        v16h bb = frag_b_pk(W1t, Cin_pad, kb, ct * 16);
        acc = __builtin_amdgcn_wmma_f32_16x16x32_f16(false, a, false, bb,
                                                     (short)0, acc, false, false);
      }
      const int n = ct * 16 + (lane & 15);
      const float bias = b1[n];
      const int rbase = rt * 16 + ((lane < 16) ? 0 : 8);
#pragma unroll
      for (int v = 0; v < 8; ++v) {
        const float x = acc[v] + bias;
        Hs[(size_t)(rbase + v) * C1 + n] = (_Float16)fmaxf(x, 0.0f);
      }
    }
  }
  __syncthreads();

  // -- layer 2 + masked max-pool over rows j < k -----------------------------
  const int nct2 = C2 >> 4;
  for (int ct = wv; ct < nct2; ct += 4) {
    const int n = ct * 16 + (lane & 15);
    const float bias = b2[n];
    float pm = -3.4e38f;
    for (int rt = 0; rt < nrt; ++rt) {
      v8f acc = {};
      for (int kb = 0; kb < C1; kb += 32) {
        v16h a  = frag_a_lds(Hs, C1, rt * 16, kb);
        v16h bb = frag_b_pk(W2t, C1, kb, ct * 16);
        acc = __builtin_amdgcn_wmma_f32_16x16x32_f16(false, a, false, bb,
                                                     (short)0, acc, false, false);
      }
      const int rbase = rt * 16 + ((lane < 16) ? 0 : 8);
#pragma unroll
      for (int v = 0; v < 8; ++v) {
        if (rbase + v < k) pm = fmaxf(pm, acc[v] + bias);
      }
    }
    const float other = __shfl_xor(pm, 16);
    pm = fmaxf(pm, other);
    if (lane < 16) outp[((size_t)b * S + s) * C2 + n] = pm;
  }
}

// ---------------------------------------------------------------------------
// (B,3,N) -> (B,N,3)
// ---------------------------------------------------------------------------
__global__ void k_transpose(const float* __restrict__ pc, float* __restrict__ xyz,
                            int N, int total) {
  const int i = blockIdx.x * blockDim.x + threadIdx.x;
  if (i >= total) return;
  const int c = i % 3, t = i / 3, n = t % N, b = t / N;
  xyz[i] = pc[((size_t)b * 3 + c) * N + n];
}

// ---------------------------------------------------------------------------
// Farthest point sampling, one block per batch. argmax first-index tiebreak.
// ---------------------------------------------------------------------------
__global__ __launch_bounds__(256) void k_fps(const float* __restrict__ xyz,
                                             int* __restrict__ fidx, int N, int S) {
  extern __shared__ float fps_sm[];
  float* px = fps_sm;
  float* py = fps_sm + N;
  float* pz = fps_sm + 2 * N;
  float* dist = fps_sm + 3 * N;
  float* rv = fps_sm + 4 * N;               // 256
  int*   ri = (int*)(fps_sm + 4 * N + 256); // 256
  __shared__ int cur;
  const int b = blockIdx.x, tid = threadIdx.x;
  const float* X = xyz + (size_t)b * N * 3;
  for (int j = tid; j < N; j += 256) {
    px[j] = X[j * 3 + 0]; py[j] = X[j * 3 + 1]; pz[j] = X[j * 3 + 2];
    dist[j] = 1e10f;
  }
  if (tid == 0) cur = 0;
  __syncthreads();
  for (int it = 0; it < S; ++it) {
    const int c = cur;
    if (tid == 0) fidx[(size_t)b * S + it] = c;
    const float cx = px[c], cy = py[c], cz = pz[c];
    float bv = -1.0f; int bi = 0x7fffffff;
    for (int j = tid; j < N; j += 256) {
      const float dx = px[j] - cx, dy = py[j] - cy, dz = pz[j] - cz;
      const float d = dx * dx + dy * dy + dz * dz;
      const float nd = fminf(dist[j], d);
      dist[j] = nd;
      if (nd > bv || (nd == bv && j < bi)) { bv = nd; bi = j; }
    }
    rv[tid] = bv; ri[tid] = bi;
    __syncthreads();
    for (int off = 128; off > 0; off >>= 1) {
      if (tid < off) {
        const float ov = rv[tid + off]; const int oi = ri[tid + off];
        if (ov > rv[tid] || (ov == rv[tid] && oi < ri[tid])) { rv[tid] = ov; ri[tid] = oi; }
      }
      __syncthreads();
    }
    if (tid == 0) cur = ri[0];
    __syncthreads();
  }
}

// ---------------------------------------------------------------------------
// gather centroid coords: out(B,S,3) = xyz[fidx]
// ---------------------------------------------------------------------------
__global__ void k_gather3(const float* __restrict__ xyz, const int* __restrict__ fidx,
                          float* __restrict__ out, int N, int S, int total) {
  const int i = blockIdx.x * blockDim.x + threadIdx.x;
  if (i >= total) return;
  const int c = i % 3, t = i / 3, s = t % S, b = t / S;
  const int p = fidx[(size_t)b * S + s];
  out[i] = xyz[((size_t)b * N + p) * 3 + c];
}

// ---------------------------------------------------------------------------
// kNN: one block per (centroid, batch); iterative argmin with tiebreak.
// ---------------------------------------------------------------------------
__global__ __launch_bounds__(128) void k_knn(const float* __restrict__ xyz,
                                             const float* __restrict__ cen,
                                             int* __restrict__ kidx,
                                             int N, int S, int k) {
  extern __shared__ float knn_sm[];
  float* d  = knn_sm;          // N
  float* rv = knn_sm + N;      // 128
  int*   ri = (int*)(knn_sm + N + 128); // 128
  const int b = blockIdx.y, s = blockIdx.x, tid = threadIdx.x;
  const float cx = cen[((size_t)b * S + s) * 3 + 0];
  const float cy = cen[((size_t)b * S + s) * 3 + 1];
  const float cz = cen[((size_t)b * S + s) * 3 + 2];
  for (int j = tid; j < N; j += 128) {
    const float dx = xyz[((size_t)b * N + j) * 3 + 0] - cx;
    const float dy = xyz[((size_t)b * N + j) * 3 + 1] - cy;
    const float dz = xyz[((size_t)b * N + j) * 3 + 2] - cz;
    d[j] = dx * dx + dy * dy + dz * dz;
  }
  __syncthreads();
  for (int t = 0; t < k; ++t) {
    float bv = 3.4e38f; int bi = 0x7fffffff;
    for (int j = tid; j < N; j += 128) {
      const float v = d[j];
      if (v < bv || (v == bv && j < bi)) { bv = v; bi = j; }
    }
    rv[tid] = bv; ri[tid] = bi;
    __syncthreads();
    for (int off = 64; off > 0; off >>= 1) {
      if (tid < off) {
        if (rv[tid + off] < rv[tid] ||
            (rv[tid + off] == rv[tid] && ri[tid + off] < ri[tid])) {
          rv[tid] = rv[tid + off]; ri[tid] = ri[tid + off];
        }
      }
      __syncthreads();
    }
    if (tid == 0) { kidx[((size_t)b * S + s) * k + t] = ri[0]; d[ri[0]] = 3.5e38f; }
    __syncthreads();
  }
}

// ---------------------------------------------------------------------------
// max over G group-partials: out(B,C) = max_g part(B,G,C)
// ---------------------------------------------------------------------------
__global__ void k_pool_reduce(const float* __restrict__ part, float* __restrict__ out,
                              int G, int C, int total) {
  const int i = blockIdx.x * blockDim.x + threadIdx.x;
  if (i >= total) return;
  const int b = i / C, c = i % C;
  float m = -3.4e38f;
  for (int g = 0; g < G; ++g) m = fmaxf(m, part[((size_t)b * G + g) * C + c]);
  out[i] = m;
}

// ---------------------------------------------------------------------------
// small dense: out(M,Nn) = X(M,K) @ W(K,Nn) + b
// ---------------------------------------------------------------------------
__global__ void k_linear(const float* __restrict__ X, const float* __restrict__ W,
                         const float* __restrict__ bias, float* __restrict__ out,
                         int M, int K, int Nn) {
  const int i = blockIdx.x * blockDim.x + threadIdx.x;
  if (i >= M * Nn) return;
  const int m = i / Nn, n = i % Nn;
  float acc = bias[n];
  for (int kk = 0; kk < K; ++kk) acc += X[(size_t)m * K + kk] * W[(size_t)kk * Nn + n];
  out[i] = acc;
}

// ---------------------------------------------------------------------------
// batch-stat BN (over 64 rows) + ReLU; one block (64 threads) per column.
// ---------------------------------------------------------------------------
__global__ __launch_bounds__(64) void k_bn_relu(const float* __restrict__ X,
                                                const float* __restrict__ g,
                                                const float* __restrict__ be,
                                                float* __restrict__ out, int H) {
  const int c = blockIdx.x, r = threadIdx.x;
  const float x = X[(size_t)r * H + c];
  __shared__ float s1[64], s2[64];
  s1[r] = x; s2[r] = x * x;
  __syncthreads();
  for (int off = 32; off > 0; off >>= 1) {
    if (r < off) { s1[r] += s1[r + off]; s2[r] += s2[r + off]; }
    __syncthreads();
  }
  const float mean = s1[0] * (1.0f / 64.0f);
  const float var = s2[0] * (1.0f / 64.0f) - mean * mean;
  const float y = g[c] * (x - mean) * rsqrtf(var + 1e-5f) + be[c];
  out[(size_t)r * H + c] = fmaxf(y, 0.0f);
}

// ---------------------------------------------------------------------------
// host orchestration
// ---------------------------------------------------------------------------
static inline int cdiv(int a, int b) { return (a + b - 1) / b; }

extern "C" void kernel_launch(void* const* d_in, const int* in_sizes, int n_in,
                              void* d_out, int out_size, void* d_ws, size_t ws_size,
                              hipStream_t stream) {
  (void)in_sizes; (void)n_in; (void)out_size; (void)ws_size;
  const int B = 64, N0 = 2048;

  const float* pc = (const float*)d_in[0];
  // flattened param order (jax pytree: sorted dict keys, tuple/list order kept)
  const float* cls1W = (const float*)d_in[1];  const float* cls1b = (const float*)d_in[2];
  const float* cls2W = (const float*)d_in[3];  const float* cls2b = (const float*)d_in[4];
  const float* cls3W = (const float*)d_in[5];  const float* cls3b = (const float*)d_in[6];
  // fc blocks: b1,b2,be1,be2,g1,g2,w1,w2
  const float* fc1_b1 = (const float*)d_in[7];  const float* fc1_b2 = (const float*)d_in[8];
  const float* fc1_be1= (const float*)d_in[9];  const float* fc1_be2= (const float*)d_in[10];
  const float* fc1_g1 = (const float*)d_in[11]; const float* fc1_g2 = (const float*)d_in[12];
  const float* fc1_w1 = (const float*)d_in[13]; const float* fc1_w2 = (const float*)d_in[14];
  const float* fc2_b1 = (const float*)d_in[15]; const float* fc2_b2 = (const float*)d_in[16];
  const float* fc2_be1= (const float*)d_in[17]; const float* fc2_be2= (const float*)d_in[18];
  const float* fc2_g1 = (const float*)d_in[19]; const float* fc2_g2 = (const float*)d_in[20];
  const float* fc2_w1 = (const float*)d_in[21]; const float* fc2_w2 = (const float*)d_in[22];
  const float* fc3_b1 = (const float*)d_in[23]; const float* fc3_b2 = (const float*)d_in[24];
  const float* fc3_be1= (const float*)d_in[25]; const float* fc3_be2= (const float*)d_in[26];
  const float* fc3_g1 = (const float*)d_in[27]; const float* fc3_g2 = (const float*)d_in[28];
  const float* fc3_w1 = (const float*)d_in[29]; const float* fc3_w2 = (const float*)d_in[30];
  // sa blocks: [(W,b),(W,b)]
  const float* sa1_W1 = (const float*)d_in[31]; const float* sa1_b1 = (const float*)d_in[32];
  const float* sa1_W2 = (const float*)d_in[33]; const float* sa1_b2 = (const float*)d_in[34];
  const float* sa12_W1= (const float*)d_in[35]; const float* sa12_b1= (const float*)d_in[36];
  const float* sa12_W2= (const float*)d_in[37]; const float* sa12_b2= (const float*)d_in[38];
  const float* sa13_W1= (const float*)d_in[39]; const float* sa13_b1= (const float*)d_in[40];
  const float* sa13_W2= (const float*)d_in[41]; const float* sa13_b2= (const float*)d_in[42];
  const float* sa22_W1= (const float*)d_in[43]; const float* sa22_b1= (const float*)d_in[44];
  const float* sa22_W2= (const float*)d_in[45]; const float* sa22_b2= (const float*)d_in[46];
  const float* sa23_W1= (const float*)d_in[47]; const float* sa23_b1= (const float*)d_in[48];
  const float* sa23_W2= (const float*)d_in[49]; const float* sa23_b2= (const float*)d_in[50];
  const float* sa32_W1= (const float*)d_in[51]; const float* sa32_b1= (const float*)d_in[52];
  const float* sa32_W2= (const float*)d_in[53]; const float* sa32_b2= (const float*)d_in[54];
  const float* sa33_W1= (const float*)d_in[55]; const float* sa33_b1= (const float*)d_in[56];
  const float* sa33_W2= (const float*)d_in[57]; const float* sa33_b2= (const float*)d_in[58];

  // workspace carve-out
  char* ws = (char*)d_ws;
  size_t off = 0;
  auto allocf = [&](size_t n) {
    float* p = (float*)(ws + off);
    off = (off + n * sizeof(float) + 255) & ~(size_t)255;
    return p;
  };
  auto alloch = [&](size_t n) {
    _Float16* p = (_Float16*)(ws + off);
    off = (off + n * sizeof(_Float16) + 255) & ~(size_t)255;
    return p;
  };
  auto alloci = [&](size_t n) {
    int* p = (int*)(ws + off);
    off = (off + n * sizeof(int) + 255) & ~(size_t)255;
    return p;
  };
  float* xyz0  = allocf((size_t)B * N0 * 3);
  float* l1x   = allocf((size_t)B * 512 * 3);
  float* l1p   = allocf((size_t)B * 512 * 128);
  float* l12x  = allocf((size_t)B * 128 * 3);
  float* l12p  = allocf((size_t)B * 128 * 256);
  float* l22x  = allocf((size_t)B * 256 * 3);
  float* l22p  = allocf((size_t)B * 256 * 256);
  float* l32x  = allocf((size_t)B * 256 * 3);
  float* l32p  = allocf((size_t)B * 256 * 256);
  float* f13in = allocf((size_t)B * 512);
  float* f23in = allocf((size_t)B * 512);
  float* f33in = allocf((size_t)B * 1024);
  float* part  = allocf((size_t)B * 16 * 1024);
  float* tmpA  = allocf((size_t)B * 512);
  float* tmpB  = allocf((size_t)B * 512);
  int*   fidx  = alloci((size_t)B * 512);
  int*   kidx  = alloci((size_t)B * 512 * 24);
  // packed f16 weights: (Nn x Kpad)
  _Float16* sa1_W1t  = alloch((size_t)64 * 32);
  _Float16* sa1_W2t  = alloch((size_t)128 * 64);
  _Float16* sa12_W1t = alloch((size_t)128 * 160);
  _Float16* sa12_W2t = alloch((size_t)256 * 128);
  _Float16* sa13_W1t = alloch((size_t)256 * 288);
  _Float16* sa13_W2t = alloch((size_t)512 * 256);
  _Float16* sa22_W1t = alloch((size_t)128 * 160);
  _Float16* sa22_W2t = alloch((size_t)256 * 128);
  _Float16* sa23_W1t = alloch((size_t)512 * 288);
  _Float16* sa23_W2t = alloch((size_t)512 * 512);
  _Float16* sa32_W1t = alloch((size_t)256 * 288);
  _Float16* sa32_W2t = alloch((size_t)256 * 256);
  _Float16* sa33_W1t = alloch((size_t)512 * 288);
  _Float16* sa33_W2t = alloch((size_t)1024 * 512);

  float* d_logit1 = (float*)d_out + 0;
  float* d_logit2 = (float*)d_out + 2560;
  float* d_logit3 = (float*)d_out + 5120;
  float* d_f13    = (float*)d_out + 7680;
  float* d_f23    = (float*)d_out + 15872;
  float* d_f33    = (float*)d_out + 24064;

  auto fps_sm = [&](int N) { return (size_t)(4 * N + 256) * 4 + 256 * 4; };
  auto knn_sm = [&](int N) { return (size_t)(N + 128) * 4 + 128 * 4; };
  auto sa_sm  = [&](int kp, int cinp, int c1) { return (size_t)(kp * cinp + kp * c1) * 2; };
  auto pack = [&](const float* W, _Float16* Wt, int K, int Nn, int Kpad) {
    const int tot = Nn * Kpad;
    k_pack_w<<<cdiv(tot, 256), 256, 0, stream>>>(W, Wt, K, Nn, Kpad, tot);
  };

  // ---- pack all MLP weights to f16 (K zero-padded, column-major) ----
  pack(sa1_W1, sa1_W1t, 6, 64, 32);      pack(sa1_W2, sa1_W2t, 64, 128, 64);
  pack(sa12_W1, sa12_W1t, 131, 128, 160); pack(sa12_W2, sa12_W2t, 128, 256, 128);
  pack(sa13_W1, sa13_W1t, 259, 256, 288); pack(sa13_W2, sa13_W2t, 256, 512, 256);
  pack(sa22_W1, sa22_W1t, 131, 128, 160); pack(sa22_W2, sa22_W2t, 128, 256, 128);
  pack(sa23_W1, sa23_W1t, 259, 512, 288); pack(sa23_W2, sa23_W2t, 512, 512, 512);
  pack(sa32_W1, sa32_W1t, 259, 256, 288); pack(sa32_W2, sa32_W2t, 256, 256, 256);
  pack(sa33_W1, sa33_W1t, 259, 512, 288); pack(sa33_W2, sa33_W2t, 512, 1024, 512);

  // point_cloud (B,3,N) -> xyz0 (B,N,3)
  { int tot = B * N0 * 3;
    k_transpose<<<cdiv(tot, 256), 256, 0, stream>>>(pc, xyz0, N0, tot); }

  // ---- SA1: fps(2048->512), knn k=12, mlp 6->64->128 ----
  k_fps<<<B, 256, fps_sm(N0), stream>>>(xyz0, fidx, N0, 512);
  { int tot = B * 512 * 3;
    k_gather3<<<cdiv(tot, 256), 256, 0, stream>>>(xyz0, fidx, l1x, N0, 512, tot); }
  k_knn<<<dim3(512, B), 128, knn_sm(N0), stream>>>(xyz0, l1x, kidx, N0, 512, 12);
  k_sa_fused<<<dim3(512, B), 128, sa_sm(16, 32, 64), stream>>>(
      0, xyz0, xyz0, l1x, kidx, sa1_W1t, sa1_b1, sa1_W2t, sa1_b2, l1p,
      512, N0, 12, 16, 3, 32, 64, 128);

  // ---- SA12: fps(512->128), knn k=8, mlp 131->128->256 ----
  k_fps<<<B, 256, fps_sm(512), stream>>>(l1x, fidx, 512, 128);
  { int tot = B * 128 * 3;
    k_gather3<<<cdiv(tot, 256), 256, 0, stream>>>(l1x, fidx, l12x, 512, 128, tot); }
  k_knn<<<dim3(128, B), 128, knn_sm(512), stream>>>(l1x, l12x, kidx, 512, 128, 8);
  k_sa_fused<<<dim3(128, B), 128, sa_sm(16, 160, 128), stream>>>(
      0, l1x, l1p, l12x, kidx, sa12_W1t, sa12_b1, sa12_W2t, sa12_b2, l12p,
      128, 512, 8, 16, 128, 160, 128, 256);

  // ---- SA13 group-all over 128 pts: mlp 259->256->512, max ----
  k_sa_fused<<<dim3(8, B), 128, sa_sm(16, 288, 256), stream>>>(
      1, l12x, l12p, nullptr, nullptr, sa13_W1t, sa13_b1, sa13_W2t, sa13_b2, part,
      8, 128, 16, 16, 256, 288, 256, 512);
  { int tot = B * 512;
    k_pool_reduce<<<cdiv(tot, 256), 256, 0, stream>>>(part, f13in, 8, 512, tot); }

  // ---- SA22: fps(512->256), knn k=16, mlp 131->128->256 ----
  k_fps<<<B, 256, fps_sm(512), stream>>>(l1x, fidx, 512, 256);
  { int tot = B * 256 * 3;
    k_gather3<<<cdiv(tot, 256), 256, 0, stream>>>(l1x, fidx, l22x, 512, 256, tot); }
  k_knn<<<dim3(256, B), 128, knn_sm(512), stream>>>(l1x, l22x, kidx, 512, 256, 16);
  k_sa_fused<<<dim3(256, B), 128, sa_sm(16, 160, 128), stream>>>(
      0, l1x, l1p, l22x, kidx, sa22_W1t, sa22_b1, sa22_W2t, sa22_b2, l22p,
      256, 512, 16, 16, 128, 160, 128, 256);

  // ---- SA23 group-all over 256 pts: mlp 259->512->512, max ----
  k_sa_fused<<<dim3(16, B), 128, sa_sm(16, 288, 512), stream>>>(
      1, l22x, l22p, nullptr, nullptr, sa23_W1t, sa23_b1, sa23_W2t, sa23_b2, part,
      16, 256, 16, 16, 256, 288, 512, 512);
  { int tot = B * 512;
    k_pool_reduce<<<cdiv(tot, 256), 256, 0, stream>>>(part, f23in, 16, 512, tot); }

  // ---- SA32: fps(256->256), knn k=24, mlp 259->256->256 ----
  k_fps<<<B, 256, fps_sm(256), stream>>>(l22x, fidx, 256, 256);
  { int tot = B * 256 * 3;
    k_gather3<<<cdiv(tot, 256), 256, 0, stream>>>(l22x, fidx, l32x, 256, 256, tot); }
  k_knn<<<dim3(256, B), 128, knn_sm(256), stream>>>(l22x, l32x, kidx, 256, 256, 24);
  k_sa_fused<<<dim3(256, B), 128, sa_sm(32, 288, 256), stream>>>(
      0, l22x, l22p, l32x, kidx, sa32_W1t, sa32_b1, sa32_W2t, sa32_b2, l32p,
      256, 256, 24, 32, 256, 288, 256, 256);

  // ---- SA33 group-all over 256 pts: mlp 259->512->1024, max ----
  k_sa_fused<<<dim3(16, B), 128, sa_sm(16, 288, 512), stream>>>(
      1, l32x, l32p, nullptr, nullptr, sa33_W1t, sa33_b1, sa33_W2t, sa33_b2, part,
      16, 256, 16, 16, 256, 288, 512, 1024);
  { int tot = B * 1024;
    k_pool_reduce<<<cdiv(tot, 256), 256, 0, stream>>>(part, f33in, 16, 1024, tot); }

  // ---- FC heads (linear -> batch BN -> relu, x2); outputs straight to d_out
  // fc1: 512 -> 256 -> 128
  k_linear<<<cdiv(B * 256, 256), 256, 0, stream>>>(f13in, fc1_w1, fc1_b1, tmpA, B, 512, 256);
  k_bn_relu<<<256, 64, 0, stream>>>(tmpA, fc1_g1, fc1_be1, tmpB, 256);
  k_linear<<<cdiv(B * 128, 256), 256, 0, stream>>>(tmpB, fc1_w2, fc1_b2, tmpA, B, 256, 128);
  k_bn_relu<<<128, 64, 0, stream>>>(tmpA, fc1_g2, fc1_be2, d_f13, 128);
  // fc2: 512 -> 256 -> 128
  k_linear<<<cdiv(B * 256, 256), 256, 0, stream>>>(f23in, fc2_w1, fc2_b1, tmpA, B, 512, 256);
  k_bn_relu<<<256, 64, 0, stream>>>(tmpA, fc2_g1, fc2_be1, tmpB, 256);
  k_linear<<<cdiv(B * 128, 256), 256, 0, stream>>>(tmpB, fc2_w2, fc2_b2, tmpA, B, 256, 128);
  k_bn_relu<<<128, 64, 0, stream>>>(tmpA, fc2_g2, fc2_be2, d_f23, 128);
  // fc3: 1024 -> 512 -> 256
  k_linear<<<cdiv(B * 512, 256), 256, 0, stream>>>(f33in, fc3_w1, fc3_b1, tmpA, B, 1024, 512);
  k_bn_relu<<<512, 64, 0, stream>>>(tmpA, fc3_g1, fc3_be1, tmpB, 512);
  k_linear<<<cdiv(B * 256, 256), 256, 0, stream>>>(tmpB, fc3_w2, fc3_b2, tmpA, B, 512, 256);
  k_bn_relu<<<256, 64, 0, stream>>>(tmpA, fc3_g2, fc3_be2, d_f33, 256);

  // ---- classifier heads ----
  k_linear<<<cdiv(B * 40, 256), 256, 0, stream>>>(d_f13, cls1W, cls1b, d_logit1, B, 128, 40);
  k_linear<<<cdiv(B * 40, 256), 256, 0, stream>>>(d_f23, cls2W, cls2b, d_logit2, B, 128, 40);
  k_linear<<<cdiv(B * 40, 256), 256, 0, stream>>>(d_f33, cls3W, cls3b, d_logit3, B, 256, 40);
}
